// ProLoNet_84722524881110
// MI455X (gfx1250) — matrix operand order repaired and checked by reading, not compiled, not run on hardware
//
#include <hip/hip_runtime.h>

typedef __attribute__((ext_vector_type(16))) _Float16 v16h;
typedef __attribute__((ext_vector_type(8)))  _Float16 v8h;
typedef __attribute__((ext_vector_type(8)))  float    v8f;
typedef __attribute__((ext_vector_type(4)))  unsigned uv4;
typedef __attribute__((ext_vector_type(8)))  unsigned uv8;

#define NODES   255
#define IN_DIM  512
#define OUT_DIM 64
#define TILE_B  128            // batch rows per block = 8 waves * 16
#define LTP     264            // f16 row pitch for L^T/R^T/AP^T (16B aligned)
#define SWP     40             // f16 row pitch for W k-slice rows

// ---- LDS layout (byte offsets) ----
#define OFF_LT   0
#define OFF_RT   (256 * LTP * 2)                 // 135168
#define OFF_APT  (2 * 256 * LTP * 2)             // 270336
#define OFF_SW   (OFF_APT + 64 * LTP * 2)        // 304128
#define SMEM_BYTES (OFF_SW + 256 * SWP * 2)      // 324608
static_assert(SMEM_BYTES <= 320 * 1024, "LDS overflow");

// ---- global f16 scratch layout (element offsets into d_ws) ----
#define WS_WH     0                              // W f16 [256][512]
#define WS_LT     (256 * 512)                    // L^T f16 [256][256]
#define WS_RT     (WS_LT + 256 * 256)
#define WS_APT    (WS_RT + 256 * 256)            // AP^T f16 [64][256]
#define WS_HALVES (WS_APT + 64 * 256)            // 278528
#define WS_BYTES  (WS_HALVES * 2)                // 557056

#if __has_builtin(__builtin_amdgcn_sched_group_barrier)
#define SGB(mask, size, id) __builtin_amdgcn_sched_group_barrier((mask), (size), (id))
#else
#define SGB(mask, size, id)
#endif
// masks: 0x008 = MFMA/WMMA, 0x100 = DS read

static __device__ inline unsigned pk2(float a, float b) {
  union { _Float16 h[2]; unsigned u; } t;
  t.h[0] = (_Float16)a; t.h[1] = (_Float16)b;
  return t.u;
}

// Build a 32x16 f16 B-operand (K=32 group, N=16 batch) from two consecutive
// C-layout f32 tiles (t0: K 0..15, t1: K 16..31) lane-local, no swap: elems 0..7 from t0, 8..15 from t1.
static __device__ inline v16h build_bT(const float* x0, const float* x1, bool lo) {
  union { v16h v; unsigned u[8]; } B;
#pragma unroll
  for (int j = 0; j < 4; ++j) {
    unsigned p0  = pk2(x0[2*j], x0[2*j+1]);
    unsigned p1  = pk2(x1[2*j], x1[2*j+1]);
    unsigned p0x = (unsigned)__shfl_xor((int)p0, 16, 32);
    unsigned p1x = (unsigned)__shfl_xor((int)p1, 16, 32);
    B.u[j]     = p0;
    B.u[4 + j] = p1;
  }
  return B.v;
}

// A-operand (16x32 f16) from an LDS row: two 8-half runs at koff, koff+16.
static __device__ inline v16h ld_a_lds(const _Float16* rowp, int koff) {
  union { v16h v; v8h h[2]; } A;
  A.h[0] = *(const v8h*)(rowp + koff);
  A.h[1] = *(const v8h*)(rowp + koff + 16);
  return A.v;
}

// Stable log-sigmoid with hardware transcendentals only (no libm log1p):
// x = e^{-|z|} in (0,1]; for x < 2^-24 the log1p error is <= x ~ 1e-7 abs.
static __device__ inline float logsig(float z) {
  return fminf(z, 0.f) - __logf(1.f + __expf(-fabsf(z)));
}

// ---- TDM: 2D tile load global(f16) -> LDS with row padding ----
// pad_int: pad after 2<<pad_int dwords; pad_amt: (pad_amt+1) dwords of pad.
static __device__ inline void tdm_load_2d(unsigned lds_off, const _Float16* gp,
                                          unsigned dim0, unsigned rows,
                                          unsigned stride0, unsigned tile0,
                                          unsigned pad_int, unsigned pad_amt) {
  unsigned long long ga = (unsigned long long)(size_t)gp;
  uv4 g0;
  g0[0] = 1u;                                        // count=1, user descriptor
  g0[1] = lds_off;                                   // LDS byte address
  g0[2] = (unsigned)ga;                              // global_addr[31:0]
  g0[3] = (unsigned)((ga >> 32) & 0x01FFFFFFull) | (2u << 30);   // addr hi | type=2
  uv8 g1;
  g1[0] = (1u << 16) | (1u << 20) | (pad_int << 22) | (pad_amt << 25); // 2B elems, pad_en
  g1[1] = (dim0 & 0xFFFFu) << 16;                    // tensor_dim0 lo16 @bit48
  g1[2] = ((dim0 >> 16) & 0xFFFFu) | ((rows & 0xFFFFu) << 16);   // dim0 hi | dim1 lo
  g1[3] = ((rows >> 16) & 0xFFFFu) | ((tile0 & 0xFFFFu) << 16);  // dim1 hi | tile_dim0
  g1[4] = rows & 0xFFFFu;                            // tile_dim1 | tile_dim2=0
  g1[5] = stride0;                                   // tensor_dim0_stride lo32
  g1[6] = 0u;
  g1[7] = 0u;
  uv4 gz = {0u, 0u, 0u, 0u};                         // groups 2/3: dims unused (2D)
  asm volatile("tensor_load_to_lds %0, %1, %2, %3"
               :: "s"(g0), "s"(g1), "s"(gz), "s"(gz)
               : "memory");
}

// ---- prep kernel: W -> f16 (row 255 clamped), L^T/R^T/AP^T -> f16 ----
__global__ void prolonet_prep(const float* __restrict__ W,  const float* __restrict__ Lm,
                              const float* __restrict__ Rm, const float* __restrict__ AP,
                              _Float16* __restrict__ wsh) {
  int idx = blockIdx.x * 256 + threadIdx.x;
  if (idx < 131072) {                               // W f16 [256][512]
    int r = idx >> 9, k = idx & 511;
    int rr = r < NODES ? r : NODES - 1;
    wsh[idx] = (_Float16)W[rr * IN_DIM + k];
  } else if (idx < 196608) {                        // L^T [leaf][node], node 255 = 0
    int j = idx - 131072;
    int leaf = j >> 8, node = j & 255;
    wsh[idx] = (_Float16)(node < NODES ? Lm[node * 256 + leaf] : 0.f);
  } else if (idx < 262144) {                        // R^T
    int j = idx - 196608;
    int leaf = j >> 8, node = j & 255;
    wsh[idx] = (_Float16)(node < NODES ? Rm[node * 256 + leaf] : 0.f);
  } else if (idx < WS_HALVES) {                     // AP^T [out][leaf]
    int j = idx - 262144;
    int o = j >> 8, leaf = j & 255;
    wsh[idx] = (_Float16)AP[leaf * OUT_DIM + o];
  }
}

template <bool USE_TDM>
__global__ __launch_bounds__(256)
void prolonet_fused(const float* __restrict__ X,  const float* __restrict__ W,
                    const float* __restrict__ C,  const float* __restrict__ ALPHA,
                    const float* __restrict__ AP, const float* __restrict__ Lm,
                    const float* __restrict__ Rm, float* __restrict__ OUT,
                    const _Float16* __restrict__ WH,  const _Float16* __restrict__ LTH,
                    const _Float16* __restrict__ RTH, const _Float16* __restrict__ APTH)
{
  extern __shared__ char smem[];
  _Float16* sLT  = (_Float16*)(smem + OFF_LT);
  _Float16* sRT  = (_Float16*)(smem + OFF_RT);
  _Float16* sAPT = (_Float16*)(smem + OFF_APT);
  _Float16* sW   = (_Float16*)(smem + OFF_SW);

  const int tid = threadIdx.x;
  const unsigned ldsbase = (unsigned)__builtin_amdgcn_groupstaticsize();

  if (USE_TDM) {
    if (tid < 32) {   // wave 0 issues the three staging DMAs
      tdm_load_2d(ldsbase + OFF_LT,  LTH,  256, 256, 256, 256, 6, 3);
      tdm_load_2d(ldsbase + OFF_RT,  RTH,  256, 256, 256, 256, 6, 3);
      tdm_load_2d(ldsbase + OFF_APT, APTH, 256,  64, 256, 256, 6, 3);
      __builtin_amdgcn_s_wait_tensorcnt(0);
    }
  } else {
    // VALU staging fallback: transpose + convert from f32 globals
#pragma unroll 1
    for (int i = 0; i < 64; ++i) {
      int idx4 = tid + i * 256;
      if (idx4 < NODES * 64) {
        int node = idx4 >> 6;
        int lf0  = (idx4 & 63) << 2;
        float4 lv = ((const float4*)Lm)[idx4];
        float4 rv = ((const float4*)Rm)[idx4];
        sLT[(lf0 + 0) * LTP + node] = (_Float16)lv.x;
        sLT[(lf0 + 1) * LTP + node] = (_Float16)lv.y;
        sLT[(lf0 + 2) * LTP + node] = (_Float16)lv.z;
        sLT[(lf0 + 3) * LTP + node] = (_Float16)lv.w;
        sRT[(lf0 + 0) * LTP + node] = (_Float16)rv.x;
        sRT[(lf0 + 1) * LTP + node] = (_Float16)rv.y;
        sRT[(lf0 + 2) * LTP + node] = (_Float16)rv.z;
        sRT[(lf0 + 3) * LTP + node] = (_Float16)rv.w;
      }
    }
    if (tid < 256) {
      sLT[tid * LTP + 255] = (_Float16)0.f;
      sRT[tid * LTP + 255] = (_Float16)0.f;
    }
#pragma unroll 1
    for (int i = 0; i < 16; ++i) {
      int idx4 = tid + i * 256;
      int leaf = idx4 >> 4;
      int o0   = (idx4 & 15) << 2;
      float4 av = ((const float4*)AP)[idx4];
      sAPT[(o0 + 0) * LTP + leaf] = (_Float16)av.x;
      sAPT[(o0 + 1) * LTP + leaf] = (_Float16)av.y;
      sAPT[(o0 + 2) * LTP + leaf] = (_Float16)av.z;
      sAPT[(o0 + 3) * LTP + leaf] = (_Float16)av.w;
    }
  }
  __syncthreads();

  const float alphav = ALPHA[0];
  const int  wave = tid >> 5;
  const int  ln   = tid & 31;
  const int  lnm  = ln & 15;
  const bool lo   = ln < 16;
  const int  b0   = blockIdx.x * TILE_B + wave * 16;

  // ================= GEMM1: Z^T = W * X^T (K=512, f32 acc) =================
  v8f zacc[16];
#pragma unroll
  for (int t = 0; t < 16; ++t) zacc[t] = (v8f){0,0,0,0,0,0,0,0};

  const float* xrow = X + (size_t)(b0 + lnm) * IN_DIM;
  const int koffW = lo ? 0 : 8;
#pragma unroll 1
  for (int kb = 0; kb < 16; ++kb) {
    // B operand: X^T (global) — issue before the barrier to hide latency
    const float* xp = xrow + kb * 32 + (lo ? 0 : 8);
    union { v16h v; _Float16 e[16]; } bx;
#pragma unroll
    for (int i = 0; i < 16; ++i) bx.e[i] = (_Float16)xp[i < 8 ? i : i + 8];

    __syncthreads();                          // previous slice fully consumed
    if (USE_TDM) {
      if (tid < 32) {
        tdm_load_2d(ldsbase + OFF_SW, WH + kb * 32, 32, 256, 512, 32, 3, 3);
        __builtin_amdgcn_s_wait_tensorcnt(0);
      }
    } else {
      // cooperative VALU staging: thread tid stages node-row tid of this slice
      const float* wp = W + (size_t)(tid < NODES ? tid : NODES - 1) * IN_DIM + kb * 32;
      union { v8h h[4]; _Float16 e[32]; } row;
#pragma unroll
      for (int i = 0; i < 32; ++i) row.e[i] = (_Float16)wp[i];
      v8h* dst = (v8h*)(sW + tid * SWP);
#pragma unroll
      for (int i = 0; i < 4; ++i) dst[i] = row.h[i];
    }
    __syncthreads();                          // slice visible to all waves

    // rotating 2-deep pipeline; sched_group_barrier enforces ds/wmma overlap
    {
      v16h aA = ld_a_lds(sW + (0 * 16 + lnm) * SWP, koffW);
      v16h aB = ld_a_lds(sW + (1 * 16 + lnm) * SWP, koffW);
#pragma unroll
      for (int mt = 0; mt < 16; ++mt) {
        v16h cur = aA;
        aA = aB;
        if (mt + 2 < 16)
          aB = ld_a_lds(sW + ((mt + 2) * 16 + lnm) * SWP, koffW);
        zacc[mt] = __builtin_amdgcn_wmma_f32_16x16x32_f16(
            false, cur, false, bx.v, (short)0, zacc[mt], false, false);
      }
      SGB(0x100, 4, 3);                       // 2 tiles of ds-reads up front
#pragma unroll
      for (int it = 0; it < 14; ++it) {
        SGB(0x008, 1, 3);                     // 1 WMMA ...
        SGB(0x100, 2, 3);                     // ... overlapped with next tile
      }
      SGB(0x008, 2, 3);                       // drain
    }
  }

  // ====== build all GEMM2 B-operands (consumes zacc incrementally) ======
  // comparators read straight from global C (1KB, L1-resident): keeps the
  // DS-read scheduling groups below clean of non-matrix LDS traffic.
  v16h BSv[8], BMv[8];
#pragma unroll
  for (int g = 0; g < 8; ++g) {
    float ls0[8], lm0[8], ls1[8], lm1[8];
#pragma unroll
    for (int v = 0; v < 8; ++v) {
      int n0 = (2 * g) * 16 + v + (lo ? 0 : 8);
      n0 = n0 < NODES ? n0 : NODES - 1;
      float z0 = (zacc[2 * g][v] - C[n0]) * alphav;
      float l0 = logsig(z0);
      ls0[v] = l0; lm0[v] = l0 - z0;          // logsig(-z) = logsig(z) - z
      int n1 = (2 * g + 1) * 16 + v + (lo ? 0 : 8);
      n1 = n1 < NODES ? n1 : NODES - 1;
      float z1 = (zacc[2 * g + 1][v] - C[n1]) * alphav;
      float l1 = logsig(z1);
      ls1[v] = l1; lm1[v] = l1 - z1;
    }
    BSv[g] = build_bT(ls0, ls1, lo);
    BMv[g] = build_bT(lm0, lm1, lo);
  }

  // ====== GEMM2 (two halves of 8 leaf tiles) + GEMM3 B-operand build ======
  v16h BPv[8];
#pragma unroll
  for (int hv = 0; hv < 2; ++hv) {
    v8f leaf[8];
#pragma unroll
    for (int t = 0; t < 8; ++t) leaf[t] = (v8f){0,0,0,0,0,0,0,0};

    // flattened (lt8, g, L/R) space, rotating 2-deep pipeline (128 steps)
    auto a_at = [&](int s) {
      int lt8 = s >> 4, g = (s >> 1) & 7, mat = s & 1;
      const _Float16* base =
          (mat ? sRT : sLT) + (((hv * 8 + lt8) * 16) + lnm) * LTP;
      return ld_a_lds(base, g * 32 + (lo ? 0 : 8));
    };
    v16h aA = a_at(0);
    v16h aB = a_at(1);
#pragma unroll
    for (int s = 0; s < 128; ++s) {
      v16h cur = aA;
      aA = aB;
      if (s + 2 < 128) aB = a_at(s + 2);
      const int lt8 = s >> 4, g = (s >> 1) & 7, mat = s & 1;
      leaf[lt8] = __builtin_amdgcn_wmma_f32_16x16x32_f16(
          false, cur, false, mat ? BMv[g] : BSv[g], (short)0, leaf[lt8],
          false, false);
    }
    // both halves share sync ID 0; group order follows program order
    SGB(0x100, 4, 0);
#pragma unroll
    for (int it = 0; it < 126; ++it) {
      SGB(0x008, 1, 0);
      SGB(0x100, 2, 0);
    }
    SGB(0x008, 2, 0);

#pragma unroll
    for (int g4 = 0; g4 < 4; ++g4) {
      float p0[8], p1[8];
#pragma unroll
      for (int v = 0; v < 8; ++v) {
        p0[v] = __expf(leaf[2 * g4][v]);
        p1[v] = __expf(leaf[2 * g4 + 1][v]);
      }
      BPv[hv * 4 + g4] = build_bT(p0, p1, lo);
    }
  }

  // ====== GEMM3: act^T = AP^T * probs^T (flattened, pipelined, 32 steps) ======
  v8f outa[4];
#pragma unroll
  for (int t = 0; t < 4; ++t) outa[t] = (v8f){0,0,0,0,0,0,0,0};
  {
    auto a3_at = [&](int s) {
      int g = s >> 2, ot = s & 3;
      return ld_a_lds(sAPT + (ot * 16 + lnm) * LTP, g * 32 + (lo ? 0 : 8));
    };
    v16h aA = a3_at(0);
    v16h aB = a3_at(1);
#pragma unroll
    for (int s = 0; s < 32; ++s) {
      v16h cur = aA;
      aA = aB;
      if (s + 2 < 32) aB = a3_at(s + 2);
      outa[s & 3] = __builtin_amdgcn_wmma_f32_16x16x32_f16(
          false, cur, false, BPv[s >> 2], (short)0, outa[s & 3], false, false);
    }
    SGB(0x100, 4, 2);
#pragma unroll
    for (int it = 0; it < 30; ++it) {
      SGB(0x008, 1, 2);
      SGB(0x100, 2, 2);
    }
    SGB(0x008, 2, 2);
  }

  // ====== softmax over 64 outputs per batch row ======
  float mx = -3.402823466e38f;
#pragma unroll
  for (int t = 0; t < 4; ++t)
#pragma unroll
    for (int v = 0; v < 8; ++v) mx = fmaxf(mx, outa[t][v]);
  mx = fmaxf(mx, __shfl_xor(mx, 16, 32));
  float ev[4][8];
  float s = 0.f;
#pragma unroll
  for (int t = 0; t < 4; ++t)
#pragma unroll
    for (int v = 0; v < 8; ++v) { ev[t][v] = __expf(outa[t][v] - mx); s += ev[t][v]; }
  s += __shfl_xor(s, 16, 32);
  const float inv = 1.f / s;

  // ====== LDS re-stage (reuse sLT region) for coalesced output ======
  __syncthreads();
  float* so = (float*)smem + wave * (16 * 68);
#pragma unroll
  for (int t = 0; t < 4; ++t)
#pragma unroll
    for (int v = 0; v < 8; ++v)
      so[lnm * 68 + t * 16 + v + (lo ? 0 : 8)] = ev[t][v] * inv;
  __syncthreads();

#pragma unroll
  for (int i = 0; i < 8; ++i) {
    int idx = i * 32 + ln;
    int r = idx >> 4, c4 = idx & 15;
    float4 val = *(const float4*)(so + r * 68 + c4 * 4);
    *(float4*)(OUT + (size_t)(b0 + r) * OUT_DIM + c4 * 4) = val;
  }
}

extern "C" void kernel_launch(void* const* d_in, const int* in_sizes, int n_in,
                              void* d_out, int out_size, void* d_ws, size_t ws_size,
                              hipStream_t stream) {
  (void)in_sizes; (void)n_in; (void)out_size;
  const float* X  = (const float*)d_in[0];
  const float* W  = (const float*)d_in[1];
  const float* C  = (const float*)d_in[2];
  const float* A  = (const float*)d_in[3];
  const float* AP = (const float*)d_in[4];
  const float* Lm = (const float*)d_in[5];
  const float* Rm = (const float*)d_in[6];
  float* OUT = (float*)d_out;

  dim3 grid(32768 / TILE_B), block(256);
  if (ws_size >= (size_t)WS_BYTES && d_ws != nullptr) {
    _Float16* wsh = (_Float16*)d_ws;
    prolonet_prep<<<dim3((WS_HALVES + 255) / 256), block, 0, stream>>>(W, Lm, Rm, AP, wsh);
    (void)hipFuncSetAttribute(reinterpret_cast<const void*>(&prolonet_fused<true>),
                              hipFuncAttributeMaxDynamicSharedMemorySize, (int)SMEM_BYTES);
    prolonet_fused<true><<<grid, block, SMEM_BYTES, stream>>>(
        X, W, C, A, AP, Lm, Rm, OUT,
        wsh + WS_WH, wsh + WS_LT, wsh + WS_RT, wsh + WS_APT);
  } else {
    (void)hipFuncSetAttribute(reinterpret_cast<const void*>(&prolonet_fused<false>),
                              hipFuncAttributeMaxDynamicSharedMemorySize, (int)SMEM_BYTES);
    prolonet_fused<false><<<grid, block, SMEM_BYTES, stream>>>(
        X, W, C, A, AP, Lm, Rm, OUT, nullptr, nullptr, nullptr, nullptr);
  }
}
